// LSTM_83176336654501
// MI455X (gfx1250) — matrix operand order repaired and checked by reading, not compiled, not run on hardware
//
#include <hip/hip_runtime.h>

// MI455X / gfx1250 persistent-LSTM kernel, round 3.
// Grid = 32 workgroups x 256 threads (8 wave32); each workgroup owns a
// 16-row batch tile and runs the full T=2048 recurrence out of LDS.
//   - h double-buffered in LDS -> ONE __syncthreads per timestep (serial
//     recurrence => per-step latency is the bottleneck)
//   - x slice staged by the Tensor Data Mover (tensor_load_to_lds, D# per
//     ISA 8.3/8.4) + s_wait_tensorcnt  [6-arg clang-23 builtin form]
//   - gates/projection: v_wmma_f32_16x16x32_f16 (20 WMMA/wave/step)

typedef __attribute__((ext_vector_type(16))) _Float16 v16h;
typedef __attribute__((ext_vector_type(8)))  _Float16 v8h;
typedef __attribute__((ext_vector_type(8)))  float    v8f;
typedef __attribute__((ext_vector_type(4)))  unsigned int u32x4;
typedef __attribute__((ext_vector_type(8)))  int      i32x8;
typedef __attribute__((ext_vector_type(4)))  int      i32x4;

#define B_   512
#define T_   2048
#define H_   128
#define O_   128
#define G_   512        // 4*H gates
#define WSTR 136        // padded LDS row stride in halves (breaks bank conflicts)
#define MTILE 16
#define NTHREADS 256

// LDS layout (bytes): [ W_hh f16 | h ping | h pong | x f32 ]
#define WHH_HALVES   (G_ * WSTR)              // 69632 halves
#define HBUF_HALVES  (MTILE * WSTR)           // 2176 halves per buffer
#define XSH_OFF      ((WHH_HALVES + 2 * HBUF_HALVES) * 2)   // 147968 bytes
#define SMEM_BYTES   (XSH_OFF + MTILE * T_ * 4)             // 279040 bytes

__device__ __forceinline__ float sigm(float x) {
    return __frcp_rn(1.0f + __expf(-x));
}
__device__ __forceinline__ float tanh_fast(float x) {
    float e = __expf(2.0f * x);
    return (e - 1.0f) * __frcp_rn(e + 1.0f);
}

__global__ __launch_bounds__(NTHREADS, 1)
void lstm_wmma_kernel(const float* __restrict__ inputs,
                      const float* __restrict__ W_ih,
                      const float* __restrict__ W_hh,
                      const float* __restrict__ b_ih,
                      const float* __restrict__ b_hh,
                      const float* __restrict__ W_lin,
                      const float* __restrict__ b_lin,
                      float* __restrict__ out)
{
    extern __shared__ char smem[];
    _Float16* whh = (_Float16*)smem;                  // [G_][WSTR] f16
    _Float16* hsh = whh + WHH_HALVES;                 // 2 x [MTILE][WSTR] f16
    float*    xsh = (float*)(smem + XSH_OFF);         // [MTILE][T_] f32 row-major

    const int tid  = threadIdx.x;
    const int wave = tid >> 5;
    const int lane = tid & 31;
    const int l15  = lane & 15;
    const int hi   = lane >> 4;              // half-wave select
    const int row0 = blockIdx.x * MTILE;     // batch tile base row

    // ------------- preamble: W_hh f32->f16 into LDS (padded rows) ---------
    for (int i = tid; i < G_ * H_; i += NTHREADS) {
        int r = i >> 7, k = i & (H_ - 1);
        whh[r * WSTR + k] = (_Float16)W_hh[i];
    }

    // ------------- preamble: x tile via Tensor Data Mover -----------------
#if __has_builtin(__builtin_amdgcn_tensor_load_to_lds)
    if (wave == 0) {
        unsigned long long ga =
            (unsigned long long)(const void*)(inputs + (size_t)row0 * T_);
        // D# group 0: count=1 | lds_addr | global_addr | type=2 ("image")
        u32x4 g0;
        g0.x = 1u;
        g0.y = (unsigned)XSH_OFF;
        g0.z = (unsigned)(ga & 0xFFFFFFFFu);
        g0.w = (unsigned)((ga >> 32) & 0x01FFFFFFu) | 0x80000000u;
        // D# group 1: data_size=4B; tensor_dim0=2048, tensor_dim1=512;
        // tile = [2048 x 16]; tensor_dim0_stride = 2048 elements
        i32x8 g1;
        g1[0] = 0x20000;                  // data_size=2 (4 bytes), no flags
        g1[1] = (int)(0x0800u << 16);     // tensor_dim0[15:0]=2048 @ bits63:48
        g1[2] = (int)(0x0200u << 16);     // tensor_dim1[15:0]=512  @ bits95:80
        g1[3] = (int)(0x0800u << 16);     // tile_dim0=2048         @ bits127:112
        g1[4] = MTILE;                    // tile_dim1=16
        g1[5] = T_;                       // tensor_dim0_stride low 32
        g1[6] = 0;
        g1[7] = 0;
        i32x4 gz4 = {0, 0, 0, 0};         // 2D tensor: groups 2/3 unused
        i32x8 gz8 = {0, 0, 0, 0, 0, 0, 0, 0};
        // clang-23 / therock-10.0 6-arg form:
        // (uint32x4 g0, int32x8 g1, int32x4, int32x4, int32x8, i32 cpol)
        __builtin_amdgcn_tensor_load_to_lds(g0, g1, gz4, gz4, gz8, 0);
        __builtin_amdgcn_s_wait_tensorcnt(0);
    }
#else
    for (int i = tid; i < MTILE * T_; i += NTHREADS) {
        int m = i >> 11, t = i & (T_ - 1);
        xsh[m * T_ + t] = inputs[(size_t)(row0 + m) * T_ + t];
    }
#endif

    // per-lane constants: hidden/output column j, gate params for i/f/g/o
    const int j = wave * 16 + l15;
    float wih[4], bias[4];
    #pragma unroll
    for (int q = 0; q < 4; ++q) {
        int n = j + q * H_;
        wih[q]  = W_ih[n];                 // D == 1 -> W_ih is a [4H] vector
        bias[q] = b_ih[n] + b_hh[n];
    }
    const float blin = b_lin[j];

    // W_lin B fragments (constant over time): B[k][n] = W_lin[n][k]
    v16h wlinB[4];
    #pragma unroll
    for (int kt = 0; kt < 4; ++kt) {
        const float* src = W_lin + j * H_ + kt * 32 + hi * 16;
        v16h b;
        #pragma unroll
        for (int e = 0; e < 16; ++e) b[e] = (_Float16)src[e];
        wlinB[kt] = b;
    }

    __syncthreads();   // W_hh + x staged, h buffers owned per-step below

    // A fragments hold h_{t-1}; h_{-1} = 0 so start from zero registers.
    v16h afrag[4];
    #pragma unroll
    for (int kt = 0; kt < 4; ++kt) { v16h z = {}; afrag[kt] = z; }

    v8f c = {};   // cell state in WMMA C-layout (m = r + 8*hi, n = j)

    #pragma unroll 1
    for (int t = 0; t < T_; ++t) {
        _Float16* hbuf = hsh + (t & 1) * HBUF_HALVES;   // this step's h_t

        // x_t for the 8 batch rows of this half-wave (broadcast ds loads)
        const float* xr = xsh + t + (size_t)(8 * hi) * T_;
        float x8[8];
        #pragma unroll
        for (int r = 0; r < 8; ++r) x8[r] = xr[r * T_];

        // acc init = x_t * w_ih + (b_ih + b_hh)
        v8f acc[4];
        #pragma unroll
        for (int q = 0; q < 4; ++q) {
            v8f a;
            #pragma unroll
            for (int r = 0; r < 8; ++r) a[r] = fmaf(x8[r], wih[q], bias[q]);
            acc[q] = a;
        }

        // gates += h_{t-1} @ W_hh^T : 16 WMMA per wave, 4 independent chains
        #pragma unroll
        for (int q = 0; q < 4; ++q) {
            const _Float16* brow = whh + (j + q * H_) * WSTR + hi * 16;
            #pragma unroll
            for (int kt = 0; kt < 4; ++kt) {
                v16h b = *(const v16h*)(brow + kt * 32);
                acc[q] = __builtin_amdgcn_wmma_f32_16x16x32_f16(
                    false, afrag[kt], false, b, (short)0, acc[q], false, false);
            }
        }

        // activations + state update; publish h_t (f16) into this step's buffer
        #pragma unroll
        for (int r = 0; r < 8; ++r) {
            float ig = sigm(acc[0][r]);
            float fg = sigm(acc[1][r]);
            float gg = tanh_fast(acc[2][r]);
            float og = sigm(acc[3][r]);
            float cv = fg * c[r] + ig * gg;
            c[r] = cv;
            float hv = og * tanh_fast(cv);
            hbuf[(r + 8 * hi) * WSTR + j] = (_Float16)hv;
        }

        __syncthreads();   // h_t visible; next step writes the OTHER buffer,
                           // so no write-after-read guard barrier is needed

        // rebuild A fragments from h_t (feeds projection AND next step's gates)
        #pragma unroll
        for (int kt = 0; kt < 4; ++kt) {
            const _Float16* ha = hbuf + l15 * WSTR + kt * 32 + hi * 16;
            v8h lo  = *(const v8h*)ha;
            v8h hi8 = *(const v8h*)(ha + 8);
            afrag[kt] = __builtin_shufflevector(
                lo, hi8, 0,1,2,3,4,5,6,7,8,9,10,11,12,13,14,15);
        }

        // projection: out_t = h_t @ W_lin^T + b_lin : 4 WMMA per wave
        v8f pacc;
        #pragma unroll
        for (int r = 0; r < 8; ++r) pacc[r] = blin;
        #pragma unroll
        for (int kt = 0; kt < 4; ++kt)
            pacc = __builtin_amdgcn_wmma_f32_16x16x32_f16(
                false, afrag[kt], false, wlinB[kt], (short)0, pacc, false, false);

        // coalesced f32 store: out[b, t, o]
        #pragma unroll
        for (int r = 0; r < 8; ++r) {
            size_t idx = ((size_t)(row0 + r + 8 * hi) * T_ + t) * O_ + j;
            out[idx] = pacc[r];
        }
    }
}

extern "C" void kernel_launch(void* const* d_in, const int* in_sizes, int n_in,
                              void* d_out, int out_size, void* d_ws, size_t ws_size,
                              hipStream_t stream) {
    const float* inputs = (const float*)d_in[0];
    const float* W_ih   = (const float*)d_in[1];
    const float* W_hh   = (const float*)d_in[2];
    const float* b_ih   = (const float*)d_in[3];
    const float* b_hh   = (const float*)d_in[4];
    const float* W_lin  = (const float*)d_in[5];
    const float* b_lin  = (const float*)d_in[6];
    float* out = (float*)d_out;

    lstm_wmma_kernel<<<dim3(B_ / MTILE), dim3(NTHREADS), (size_t)SMEM_BYTES,
                       stream>>>(inputs, W_ih, W_hh, b_ih, b_hh, W_lin, b_lin, out);
}
